// NonLocalBlock_5308579578210
// MI455X (gfx1250) — matrix operand order repaired and checked by reading, not compiled, not run on hardware
//
#include <hip/hip_runtime.h>

#define BATCH 4
#define CCH   256
#define HIDD  128
#define NPIX  4096   // 64*64

typedef __attribute__((ext_vector_type(16))) __bf16          v16bf;
typedef __attribute__((ext_vector_type(8)))  float           v8f;
typedef __attribute__((ext_vector_type(16))) unsigned short  v16u;
typedef __attribute__((ext_vector_type(8)))  unsigned short  u16x8;

#define HAVE_ASYNC_COPY 1

#if HAVE_ASYNC_COPY
// async DMA of 16B global -> LDS (per lane), tracked by ASYNCcnt.
// vdst = 32-bit LDS byte address (generic LDS ptr low 32 bits, per ISA aperture rules),
// vaddr = 64-bit global address.
__device__ __forceinline__ void async_cp16(const void* g, void* l) {
  unsigned lds = (unsigned)(unsigned long long)l;
  asm volatile("global_load_async_to_lds_b128 %0, %1, off"
               :: "v"(lds), "v"(g) : "memory");
}
__device__ __forceinline__ void async_wait0() {
#if defined(__has_builtin)
#if __has_builtin(__builtin_amdgcn_s_wait_asynccnt)
  __builtin_amdgcn_s_wait_asynccnt(0);
#else
  asm volatile("s_wait_asynccnt 0" ::: "memory");
#endif
#else
  asm volatile("s_wait_asynccnt 0" ::: "memory");
#endif
}
#endif

__device__ __forceinline__ unsigned short f2bf(float f) {
  __bf16 h = (__bf16)f;                       // native v_cvt -> bf16
  return __builtin_bit_cast(unsigned short, h);
}

__device__ __forceinline__ v8f wmma_bf16(v16u a, v16u b, v8f c) {
  union U { v16u u; v16bf b; } A, B;
  A.u = a; B.u = b;
  return __builtin_amdgcn_wmma_f32_16x16x32_bf16(
      false, A.b, false, B.b, (short)0, c, false, false);
}

// A fragment (16xK, row-major src, K contiguous), bf16 source.
// lanes 0-15: K = k*32 + {0..7, 16..23}; lanes 16-31: +8
__device__ __forceinline__ v16u load_afrag_bf16(const unsigned short* rowptr, int k, int hi) {
  int kb = k * 32 + (hi ? 8 : 0);
  u16x8 c0 = *(const u16x8*)(rowptr + kb);
  u16x8 c1 = *(const u16x8*)(rowptr + kb + 16);
  v16u r;
#pragma unroll
  for (int i = 0; i < 8; ++i) { r[i] = c0[i]; r[i + 8] = c1[i]; }
  return r;
}

// B fragment (Kx16): lane = column; lanes 0-15: K = k*32+{0..7, 16..23}; lanes 16-31: +8
__device__ __forceinline__ v16u load_bfrag_bf16(const unsigned short* colptr, int k, int hi) {
  int kb = k * 32 + (hi ? 8 : 0);
  u16x8 c0 = *(const u16x8*)(colptr + kb);
  u16x8 c1 = *(const u16x8*)(colptr + kb + 16);
  v16u r;
#pragma unroll
  for (int i = 0; i < 8; ++i) { r[i] = c0[i]; r[i + 8] = c1[i]; }
  return r;
}

// ---------------- Pass -1: convert a weight matrix f32 -> bf16 (one shot) ----------------
__global__ __launch_bounds__(256) void cvt_kernel(const float* __restrict__ src,
                                                  unsigned short* __restrict__ dst, int n) {
  int i = blockIdx.x * 256 + threadIdx.x;
  if (i < n) dst[i] = f2bf(src[i]);
}

// ---------------- Pass 0: projections ----------------
// theta[b][n][o], phiT[b][m][k] (= phi[k][m]) : [N,HID]; gT[b][o][n] : [HID,N]; bf16
template <bool TRANSPOSED_STORE>
__device__ __forceinline__ void proj_one(const unsigned short* __restrict__ Wb,  // [HID, C] bf16
                                         unsigned short* __restrict__ dst,
                                         const v16u* afrag, int n0, int col, int hi) {
  for (int ot = 0; ot < 8; ++ot) {
    int ocol = ot * 16 + col;
    const unsigned short* wrow = Wb + (size_t)ocol * CCH;   // B column o = W row o
    v8f acc;
#pragma unroll
    for (int i = 0; i < 8; ++i) acc[i] = 0.f;
#pragma unroll
    for (int k = 0; k < 8; ++k)
      acc = wmma_bf16(afrag[k], load_bfrag_bf16(wrow, k, hi), acc);
    if (!TRANSPOSED_STORE) {  // [N, HID] layout (scattered b16 stores)
#pragma unroll
      for (int i = 0; i < 8; ++i)
        dst[(size_t)(n0 + i + hi * 8) * HIDD + ocol] = f2bf(acc[i]);
    } else {                  // gT: [HID, N] layout -> 8 consecutive n, one b128 store
      u16x8 p;
#pragma unroll
      for (int i = 0; i < 8; ++i) p[i] = f2bf(acc[i]);
      *(u16x8*)(dst + (size_t)ocol * NPIX + n0 + hi * 8) = p;
    }
  }
}

__global__ __launch_bounds__(32) void proj_kernel(
    const float* __restrict__ x,
    const unsigned short* __restrict__ wb_theta, const unsigned short* __restrict__ wb_phi,
    const unsigned short* __restrict__ wb_g,
    unsigned short* __restrict__ theta, unsigned short* __restrict__ phiT,
    unsigned short* __restrict__ gT) {
  __shared__ __attribute__((aligned(16))) unsigned short xT[16][CCH + 8];
  int blk = blockIdx.x;
  int b = blk >> 8, nt = blk & 255, n0 = nt * 16;
  int lane = threadIdx.x, col = lane & 15, hi = lane >> 4;
  const float* xb = x + (size_t)b * CCH * NPIX;

  // load x tile [256 c x 16 n] as float4, transpose into LDS as xT[n][c] (bf16)
  for (int it = 0; it < 32; ++it) {
    int idx = it * 32 + lane;            // 1024 float4 chunks
    int c = idx >> 2, nq = (idx & 3) * 4;
    float4 v = *(const float4*)(xb + (size_t)c * NPIX + n0 + nq);
    xT[nq + 0][c] = f2bf(v.x);
    xT[nq + 1][c] = f2bf(v.y);
    xT[nq + 2][c] = f2bf(v.z);
    xT[nq + 3][c] = f2bf(v.w);
  }
  __syncthreads();

  // 8 A fragments (K = C = 256), rows n, from LDS
  v16u afrag[8];
#pragma unroll
  for (int k = 0; k < 8; ++k) {
    int kb = k * 32 + (hi ? 8 : 0);
    v16u r;
#pragma unroll
    for (int i = 0; i < 8; ++i) { r[i] = xT[col][kb + i]; r[i + 8] = xT[col][kb + 16 + i]; }
    afrag[k] = r;
  }

  size_t nh = (size_t)b * NPIX * HIDD;
  proj_one<false>(wb_theta, theta + nh, afrag, n0, col, hi);
  proj_one<false>(wb_phi,   phiT  + nh, afrag, n0, col, hi);
  proj_one<true >(wb_g,     gT    + nh, afrag, n0, col, hi);
}

// ---------------- Pass 1: softmax column stats (axis=1 => over n, per column m) ----------------
// 4 waves per WG: shared theta tile staged in LDS (double buffered), each wave owns 16 columns m.
__global__ __launch_bounds__(128) void stats_kernel(
    const unsigned short* __restrict__ theta, const unsigned short* __restrict__ phiT,
    float* __restrict__ colmax, float* __restrict__ colsum) {
  __shared__ __attribute__((aligned(16))) unsigned short thLds[2][16][HIDD + 8];
  int blk = blockIdx.x;
  int b = blk >> 6, mg = blk & 63;
  int tid = threadIdx.x, wave = tid >> 5, lane = tid & 31;
  int col = lane & 15, hi = (lane >> 4) & 1;
  int m0 = mg * 64 + wave * 16;
  const unsigned short* th = theta + (size_t)b * NPIX * HIDD;
  const unsigned short* ph = phiT  + (size_t)b * NPIX * HIDD;

  v16u bfrag[4];
  const unsigned short* prow = ph + (size_t)(m0 + col) * HIDD;
#pragma unroll
  for (int k = 0; k < 4; ++k) bfrag[k] = load_bfrag_bf16(prow, k, hi);

  auto stage = [&](int buf, int nt) {
#pragma unroll
    for (int r = 0; r < 2; ++r) {
      int idx = r * 128 + tid;                 // 256 chunks of 8 ushort
      int row = idx >> 4, koff = (idx & 15) * 8;
#if HAVE_ASYNC_COPY
      async_cp16(th + (size_t)(nt * 16 + row) * HIDD + koff, &thLds[buf][row][koff]);
#else
      *(u16x8*)&thLds[buf][row][koff] =
          *(const u16x8*)(th + (size_t)(nt * 16 + row) * HIDD + koff);
#endif
    }
  };

  float rmax = -3.4e38f, rsum = 0.f;
  stage(0, 0);
#if HAVE_ASYNC_COPY
  async_wait0();
#endif
  __syncthreads();
  for (int nt = 0; nt < NPIX / 16; ++nt) {
    int cur = nt & 1;
    if (nt + 1 < NPIX / 16) stage(1 - cur, nt + 1);
    v8f s;
#pragma unroll
    for (int i = 0; i < 8; ++i) s[i] = 0.f;
#pragma unroll
    for (int k = 0; k < 4; ++k) {
      int kb = k * 32 + (hi ? 8 : 0);
      v16u a;
#pragma unroll
      for (int i = 0; i < 8; ++i) { a[i] = thLds[cur][col][kb + i]; a[i + 8] = thLds[cur][col][kb + 16 + i]; }
      s = wmma_bf16(a, bfrag[k], s);
    }
    float tm = s[0];
#pragma unroll
    for (int i = 1; i < 8; ++i) tm = fmaxf(tm, s[i]);
    float nm = fmaxf(rmax, tm);
    float ts = 0.f;
#pragma unroll
    for (int i = 0; i < 8; ++i) ts += __expf(s[i] - nm);
    rsum = rsum * __expf(rmax - nm) + ts;
    rmax = nm;
#if HAVE_ASYNC_COPY
    async_wait0();
#endif
    __syncthreads();
  }
  float omax = __shfl_xor(rmax, 16);
  float osum = __shfl_xor(rsum, 16);
  float gm = fmaxf(rmax, omax);
  float gs = rsum * __expf(rmax - gm) + osum * __expf(omax - gm);
  if (hi == 0) {
    colmax[b * NPIX + m0 + col] = gm;
    colsum[b * NPIX + m0 + col] = gs;
  }
}

// ---------------- Pass 2: out[n,o] = sum_m softmax(s)[n,m] * g[m,o] ----------------
// 4 waves per WG covering 64 n rows; phi & gT tiles staged in LDS (double buffered).
__global__ __launch_bounds__(128) void attn_kernel(
    const unsigned short* __restrict__ theta, const unsigned short* __restrict__ phiT,
    const unsigned short* __restrict__ gT,
    const float* __restrict__ colmax, const float* __restrict__ colsum,
    unsigned short* __restrict__ attn) {
  __shared__ __attribute__((aligned(16))) unsigned short phiLds[2][32][HIDD + 8];
  __shared__ __attribute__((aligned(16))) unsigned short gLds[2][HIDD][32 + 8];
  __shared__ __attribute__((aligned(16))) unsigned short pLds[4][32][16];   // per-wave
  int blk = blockIdx.x;
  int b = blk >> 6, ng = blk & 63;
  int tid = threadIdx.x, wave = tid >> 5, lane = tid & 31;
  int col = lane & 15, hi = (lane >> 4) & 1;
  int n0 = ng * 64 + wave * 16;
  const unsigned short* th = theta + (size_t)b * NPIX * HIDD;
  const unsigned short* ph = phiT  + (size_t)b * NPIX * HIDD;
  const unsigned short* gt = gT    + (size_t)b * HIDD * NPIX;
  const float* cmax = colmax + b * NPIX;
  const float* csum = colsum + b * NPIX;

  v16u afrag[4];
  const unsigned short* trow = th + (size_t)(n0 + col) * HIDD;
#pragma unroll
  for (int k = 0; k < 4; ++k) afrag[k] = load_afrag_bf16(trow, k, hi);

  v8f acc[8];
#pragma unroll
  for (int ot = 0; ot < 8; ++ot)
#pragma unroll
    for (int i = 0; i < 8; ++i) acc[ot][i] = 0.f;

  auto stage = [&](int buf, int m0) {
#pragma unroll
    for (int r = 0; r < 4; ++r) {            // phi tile: 32 rows x 128 k = 512 chunks
      int idx = r * 128 + tid;
      int row = idx >> 4, koff = (idx & 15) * 8;
#if HAVE_ASYNC_COPY
      async_cp16(ph + (size_t)(m0 + row) * HIDD + koff, &phiLds[buf][row][koff]);
#else
      *(u16x8*)&phiLds[buf][row][koff] =
          *(const u16x8*)(ph + (size_t)(m0 + row) * HIDD + koff);
#endif
    }
#pragma unroll
    for (int r = 0; r < 4; ++r) {            // gT tile: 128 rows o x 32 m = 512 chunks
      int idx = r * 128 + tid;
      int o = idx >> 2, moff = (idx & 3) * 8;
#if HAVE_ASYNC_COPY
      async_cp16(gt + (size_t)o * NPIX + m0 + moff, &gLds[buf][o][moff]);
#else
      *(u16x8*)&gLds[buf][o][moff] =
          *(const u16x8*)(gt + (size_t)o * NPIX + m0 + moff);
#endif
    }
  };

  stage(0, 0);
#if HAVE_ASYNC_COPY
  async_wait0();
#endif
  __syncthreads();
  for (int it = 0; it < NPIX / 32; ++it) {
    int cur = it & 1;
    int m0 = it * 32;
    if (it + 1 < NPIX / 32) stage(1 - cur, m0 + 32);
    // scores + softmax for two 16-wide m tiles -> pLds[wave]
#pragma unroll
    for (int t = 0; t < 2; ++t) {
      int mloc = t * 16;
      v8f s;
#pragma unroll
      for (int i = 0; i < 8; ++i) s[i] = 0.f;
#pragma unroll
      for (int k = 0; k < 4; ++k) {
        int kb = k * 32 + (hi ? 8 : 0);
        const unsigned short* pr = &phiLds[cur][mloc + col][kb];
        u16x8 c0 = *(const u16x8*)pr;
        u16x8 c1 = *(const u16x8*)(pr + 16);
        v16u bfr;
#pragma unroll
        for (int i = 0; i < 8; ++i) { bfr[i] = c0[i]; bfr[i + 8] = c1[i]; }
        s = wmma_bf16(afrag[k], bfr, s);
      }
      float cm = cmax[m0 + mloc + col];
      float ci = 1.0f / csum[m0 + mloc + col];
      u16x8 p;
#pragma unroll
      for (int i = 0; i < 8; ++i) p[i] = f2bf(__expf(s[i] - cm) * ci);
      *(u16x8*)&pLds[wave][mloc + col][hi * 8] = p;   // same-wave LDS: in-order, no barrier
    }
    // rebuild p as A fragment [16 n x 32 m(K)]
    v16u pa;
    {
      int mb = hi * 8;
#pragma unroll
      for (int i = 0; i < 8; ++i) { pa[i] = pLds[wave][mb + i][col]; pa[i + 8] = pLds[wave][16 + mb + i][col]; }
    }
    // out += p * g   (B fragments from gLds[o][m])
#pragma unroll
    for (int ot = 0; ot < 8; ++ot) {
      int kb = hi ? 8 : 0;
      const unsigned short* gr = &gLds[cur][ot * 16 + col][kb];
      u16x8 c0 = *(const u16x8*)gr;
      u16x8 c1 = *(const u16x8*)(gr + 16);
      v16u bfr;
#pragma unroll
      for (int i = 0; i < 8; ++i) { bfr[i] = c0[i]; bfr[i + 8] = c1[i]; }
      acc[ot] = wmma_bf16(pa, bfr, acc[ot]);
    }
#if HAVE_ASYNC_COPY
    async_wait0();
#endif
    __syncthreads();
  }
  unsigned short* ao = attn + (size_t)b * NPIX * HIDD;
#pragma unroll
  for (int ot = 0; ot < 8; ++ot)
#pragma unroll
    for (int i = 0; i < 8; ++i)
      ao[(size_t)(n0 + i + hi * 8) * HIDD + ot * 16 + col] = f2bf(acc[ot][i]);
}

// ---------------- Pass 3: y = x + w_mask @ out^T ----------------
__global__ __launch_bounds__(32) void mask_kernel(
    const float* __restrict__ x, const unsigned short* __restrict__ wb_mask,
    const unsigned short* __restrict__ attn, float* __restrict__ y) {
  int blk = blockIdx.x;
  int b = blk >> 12;
  int ct = (blk >> 8) & 15, nt = blk & 255;
  int c0 = ct * 16, n0 = nt * 16;
  int lane = threadIdx.x, col = lane & 15, hi = lane >> 4;
  const unsigned short* wrow = wb_mask + (size_t)(c0 + col) * HIDD;                        // A row c
  const unsigned short* arow = attn + (size_t)b * NPIX * HIDD + (size_t)(n0 + col) * HIDD; // B col n
  v8f acc;
#pragma unroll
  for (int i = 0; i < 8; ++i) acc[i] = 0.f;
#pragma unroll
  for (int k = 0; k < 4; ++k)
    acc = wmma_bf16(load_afrag_bf16(wrow, k, hi), load_bfrag_bf16(arow, k, hi), acc);
  const float* xb = x + (size_t)b * CCH * NPIX;
  float* yb = y + (size_t)b * CCH * NPIX;
#pragma unroll
  for (int i = 0; i < 8; ++i) {
    size_t off = (size_t)(c0 + i + hi * 8) * NPIX + n0 + col;
    yb[off] = xb[off] + acc[i];
  }
}

extern "C" void kernel_launch(void* const* d_in, const int* in_sizes, int n_in,
                              void* d_out, int out_size, void* d_ws, size_t ws_size,
                              hipStream_t stream) {
  const float* x       = (const float*)d_in[0];
  const float* w_theta = (const float*)d_in[1];
  const float* w_phi   = (const float*)d_in[2];
  const float* w_g     = (const float*)d_in[3];
  const float* w_mask  = (const float*)d_in[4];
  float* out = (float*)d_out;

  char* ws = (char*)d_ws;
  size_t off = 0;
  unsigned short* theta = (unsigned short*)(ws + off); off += (size_t)BATCH * NPIX * HIDD * 2;
  unsigned short* phiT  = (unsigned short*)(ws + off); off += (size_t)BATCH * NPIX * HIDD * 2;
  unsigned short* gT    = (unsigned short*)(ws + off); off += (size_t)BATCH * HIDD * NPIX * 2;
  unsigned short* attn  = (unsigned short*)(ws + off); off += (size_t)BATCH * NPIX * HIDD * 2;
  float* colmax = (float*)(ws + off); off += (size_t)BATCH * NPIX * 4;
  float* colsum = (float*)(ws + off); off += (size_t)BATCH * NPIX * 4;
  const int WN = HIDD * CCH;  // 32768, all four weights same element count
  unsigned short* wb_theta = (unsigned short*)(ws + off); off += (size_t)WN * 2;
  unsigned short* wb_phi   = (unsigned short*)(ws + off); off += (size_t)WN * 2;
  unsigned short* wb_g     = (unsigned short*)(ws + off); off += (size_t)WN * 2;
  unsigned short* wb_mask  = (unsigned short*)(ws + off); off += (size_t)WN * 2;

  cvt_kernel<<<WN / 256, 256, 0, stream>>>(w_theta, wb_theta, WN);
  cvt_kernel<<<WN / 256, 256, 0, stream>>>(w_phi,   wb_phi,   WN);
  cvt_kernel<<<WN / 256, 256, 0, stream>>>(w_g,     wb_g,     WN);
  cvt_kernel<<<WN / 256, 256, 0, stream>>>(w_mask,  wb_mask,  WN);

  proj_kernel <<<BATCH * (NPIX / 16), 32, 0, stream>>>(x, wb_theta, wb_phi, wb_g, theta, phiT, gT);
  stats_kernel<<<BATCH * (NPIX / 64), 128, 0, stream>>>(theta, phiT, colmax, colsum);
  attn_kernel <<<BATCH * (NPIX / 64), 128, 0, stream>>>(theta, phiT, gT, colmax, colsum, attn);
  mask_kernel <<<BATCH * (CCH / 16) * (NPIX / 16), 32, 0, stream>>>(x, wb_mask, attn, out);
}
